// ThreeVoxelKernel_70884140253248
// MI455X (gfx1250) — matrix operand order repaired and checked by reading, not compiled, not run on hardware
//
#include <hip/hip_runtime.h>
#include <hip/hip_bf16.h>
#include <math.h>

// ---------------------------------------------------------------------------
// Submanifold sparse-conv tower for MI455X (gfx1250, wave32, WMMA).
// Heavy GEMMs run as bf16 v_wmma_f32_16x16x32_bf16 with f32 accumulation.
// Rulebook is transposed to k-major so index fetches are one cacheline/wave.
// ---------------------------------------------------------------------------

typedef __attribute__((ext_vector_type(16))) __bf16 v16bf;
typedef __attribute__((ext_vector_type(8)))  __bf16 v8bf;
typedef __attribute__((ext_vector_type(8)))  float  v8f;

#define CIN    3
#define CM     32
#define KK     27
#define NCLASS 20
#define NBSTAT 512

__device__ __forceinline__ unsigned short f2bf(float f) {
  unsigned int u = __float_as_uint(f);
  u += 0x7fffu + ((u >> 16) & 1u);           // round-to-nearest-even
  return (unsigned short)(u >> 16);
}

// --- Rulebook transpose: nbr[v][k] -> nbrT[k][v] (coalesced conv reads) ----
__global__ void nbr_transpose_kernel(const int* __restrict__ nbr,
                                     int* __restrict__ nbrT, int n) {
  int i = blockIdx.x * 256 + threadIdx.x;    // i = k*n + v
  if (i >= n * KK) return;
  int k = i / n;
  int v = i - k * n;
  nbrT[i] = nbr[(size_t)v * KK + k];
}

// --- Repack conv weights [4][27][32][32] (kin,cout) into WMMA B-operand ----
// layout [conv][k][half][lane][16] bf16:
//   lane n<16  holds W[k][0:8,16:24][h*16 + n]     (K=0..7,16..23)
//   lane n>=16 holds W[k][8:16,24:32][h*16 + n-16] (K=8..15,24..31)
__global__ void repack_w_kernel(const float* __restrict__ bw,
                                unsigned short* __restrict__ wpk) {
  int i = blockIdx.x * 256 + threadIdx.x;    // total 4*27*2*32*16 = 110592
  if (i >= 4 * KK * 2 * 32 * 16) return;
  int e    = i & 15;
  int lane = (i >> 4) & 31;
  int h    = (i >> 9) & 1;
  int k    = (i >> 10) % KK;
  int c    = (i >> 10) / KK;
  int kin  = (e < 8) ? ((lane >> 4) * 8 + e) : (16 + (lane >> 4) * 8 + (e - 8));
  int cout = h * 16 + (lane & 15);
  wpk[i] = f2bf(bw[((c * KK + k) * CM + kin) * CM + cout]);
}

// --- Initial SubmanifoldConvolution(3 -> 32), K=81: plain VALU -------------
__global__ void conv0_kernel(const float* __restrict__ feats,
                             const int* __restrict__ nbr,
                             const float* __restrict__ w0,
                             float* __restrict__ outF, int n) {
  __shared__ float sw[KK * CIN * CM];        // 2592 floats
  for (int i = threadIdx.x; i < KK * CIN * CM; i += blockDim.x) sw[i] = w0[i];
  __syncthreads();
  int v = blockIdx.x * blockDim.x + threadIdx.x;
  if (v >= n) return;
  float acc[CM];
#pragma unroll
  for (int o = 0; o < CM; o++) acc[o] = 0.f;
  for (int k = 0; k < KK; k++) {
    int idx = nbr[(size_t)v * KK + k];
    if (idx < n) {
      float f0 = feats[idx * 3 + 0], f1 = feats[idx * 3 + 1], f2 = feats[idx * 3 + 2];
      const float* w = &sw[k * CIN * CM];
#pragma unroll
      for (int o = 0; o < CM; o++)
        acc[o] += f0 * w[o] + f1 * w[CM + o] + f2 * w[2 * CM + o];
    }
  }
#pragma unroll
  for (int o = 0; o < CM; o++) outF[(size_t)v * CM + o] = acc[o];
}

// --- BN stage 1: fixed-order per-block partial sums (deterministic) --------
__global__ void bn_stats_kernel(const float* __restrict__ x,
                                float* __restrict__ part, int n) {
  __shared__ float ssum[256], ssq[256];
  int t = threadIdx.x;
  int ch = t & 31, g = t >> 5;
  float s = 0.f, q = 0.f;
  for (int r = blockIdx.x * 8 + g; r < n; r += NBSTAT * 8) {
    float v = x[(size_t)r * CM + ch];
    s += v; q += v * v;
  }
  ssum[t] = s; ssq[t] = q;
  __syncthreads();
  for (int off = 128; off >= 32; off >>= 1) {
    if (t < off) { ssum[t] += ssum[t + off]; ssq[t] += ssq[t + off]; }
    __syncthreads();
  }
  if (t < 32) {
    part[blockIdx.x * 64 + t]      = ssum[t];
    part[blockIdx.x * 64 + 32 + t] = ssq[t];
  }
}

// --- BN stage 2: fold partials, produce per-channel scale/shift ------------
__global__ void bn_finalize_kernel(const float* __restrict__ part,
                                   const float* __restrict__ g,
                                   const float* __restrict__ b,
                                   float* __restrict__ ss, int n) {
  int ch = threadIdx.x;
  if (ch >= 32) return;
  float s = 0.f, q = 0.f;
  for (int i = 0; i < NBSTAT; i++) {
    s += part[i * 64 + ch];
    q += part[i * 64 + 32 + ch];
  }
  float inv = 1.0f / (float)n;
  float mu  = s * inv;
  float var = q * inv - mu * mu;
  float sc  = g[ch] * rsqrtf(var + 1e-4f);
  ss[ch]      = sc;
  ss[32 + ch] = b[ch] - mu * sc;
}

// --- BN apply + ReLU; emit packed bf16 pairs (zero sentinel row N) ---------
__global__ void bn_apply_kernel(const float* __restrict__ x,
                                const float* __restrict__ ss,
                                unsigned int* __restrict__ actB2,
                                float* __restrict__ actF, int n) {
  int i = blockIdx.x * 256 + threadIdx.x;    // one bf16 pair per thread
  if (i >= (n + 1) * (CM / 2)) return;
  int e = i * 2;
  unsigned int packed = 0;
  if (e < n * CM) {
    int ch = e & 31;                         // even
    float y0 = x[e]     * ss[ch]     + ss[32 + ch];
    float y1 = x[e + 1] * ss[ch + 1] + ss[33 + ch];
    y0 = y0 > 0.f ? y0 : 0.f;
    y1 = y1 > 0.f ? y1 : 0.f;
    if (actF) { actF[e] = y0; actF[e + 1] = y1; }
    packed = (unsigned int)f2bf(y0) | ((unsigned int)f2bf(y1) << 16);
  }
  actB2[i] = packed;
}

// --- Gather-GEMM: out[16v x 32c] = sum_k A_k[16x32] @ W_k[32x32] -----------
// One wave per tile; A operand layout (16-bit A 16x32):
//   lane L<16:  row M=L,    elems 0..7 = K0..7,  elems 8..15 = K16..23
//   lane L>=16: row M=L-16, elems 0..7 = K8..15, elems 8..15 = K24..31
__global__ void __launch_bounds__(256) conv_wmma_kernel(
    const unsigned short* __restrict__ act,   // (n+1) x 32 bf16, row 'n' = zeros
    const int* __restrict__ nbrT,             // 27 x n (k-major rulebook)
    const unsigned short* __restrict__ wpk,   // 27 x 2 x 32 x 16 bf16 (packed B)
    const float* __restrict__ res,            // residual or nullptr
    float* __restrict__ out, int n) {
  int lane = threadIdx.x & 31;
  int wave = threadIdx.x >> 5;
  int base = blockIdx.x * 128 + wave * 16;
  int row  = lane & 15;
  int vr   = base + row;
  int vcl  = vr < n ? vr : (n - 1);
  bool valid = vr < n;
  int koff = (lane >> 4) * 8;

  v8f acc0 = {};
  v8f acc1 = {};
  for (int k = 0; k < KK; k++) {
    int idx = nbrT[(size_t)k * n + vcl];      // 16 consecutive dwords per wave
    idx = valid ? idx : n;                    // OOB tile rows -> zero row
    const unsigned short* arow = act + (size_t)idx * CM + koff;
    v8bf alo = *(const v8bf*)(arow);          // K[0:8]   (or K[8:16])
    v8bf ahi = *(const v8bf*)(arow + 16);     // K[16:24] (or K[24:32])
    v16bf a = __builtin_shufflevector(alo, ahi,
        0, 1, 2, 3, 4, 5, 6, 7, 8, 9, 10, 11, 12, 13, 14, 15);
    const unsigned short* bp = wpk + ((size_t)k * 2 * 32 + lane) * 16;
    v16bf b0 = *(const v16bf*)(bp);           // out channels  0..15
    v16bf b1 = *(const v16bf*)(bp + 32 * 16); // out channels 16..31
    acc0 = __builtin_amdgcn_wmma_f32_16x16x32_bf16(
        false, a, false, b0, (short)0, acc0, false, false);
    acc1 = __builtin_amdgcn_wmma_f32_16x16x32_bf16(
        false, a, false, b1, (short)0, acc1, false, false);
  }
  // C/D layout: VGPR r -> M = r (lanes 0-15) / 8+r (lanes 16-31); N = lane&15
  int col   = lane & 15;
  int mbase = base + ((lane >> 4) << 3);
#pragma unroll
  for (int r = 0; r < 8; r++) {
    int m = mbase + r;
    if (m < n) {
      float r0 = acc0[r], r1 = acc1[r];
      if (res) { r0 += res[(size_t)m * CM + col]; r1 += res[(size_t)m * CM + 16 + col]; }
      out[(size_t)m * CM + col]      = r0;
      out[(size_t)m * CM + 16 + col] = r1;
    }
  }
}

// --- Dense heads: y[20], embedding[32], sigmoid offset[1] ------------------
__global__ void heads_kernel(const float* __restrict__ feat,
    const float* __restrict__ wcls, const float* __restrict__ bcls,
    const float* __restrict__ wfe,  const float* __restrict__ bfe,
    const float* __restrict__ wle,  const float* __restrict__ ble,
    const float* __restrict__ wfr,  const float* __restrict__ bfr,
    const float* __restrict__ wlr,  const float* __restrict__ blr,
    float* __restrict__ out, int n) {
  __shared__ float s_wcls[CM * NCLASS], s_wfe[CM * CM], s_wle[CM * CM], s_wfr[CM * CM];
  __shared__ float s_wlr[CM], s_bcls[NCLASS], s_bfe[CM], s_ble[CM], s_bfr[CM], s_blr;
  int t = threadIdx.x;
  for (int i = t; i < CM * NCLASS; i += blockDim.x) s_wcls[i] = wcls[i];
  for (int i = t; i < CM * CM; i += blockDim.x) {
    s_wfe[i] = wfe[i]; s_wle[i] = wle[i]; s_wfr[i] = wfr[i];
  }
  if (t < CM) { s_wlr[t] = wlr[t]; s_bfe[t] = bfe[t]; s_ble[t] = ble[t]; s_bfr[t] = bfr[t]; }
  if (t < NCLASS) s_bcls[t] = bcls[t];
  if (t == 0) s_blr = blr[0];
  __syncthreads();
  int v = blockIdx.x * blockDim.x + t;
  if (v >= n) return;
  float f[CM];
#pragma unroll
  for (int c = 0; c < CM; c++) f[c] = feat[(size_t)v * CM + c];
#pragma unroll
  for (int o = 0; o < NCLASS; o++) {
    float a = s_bcls[o];
#pragma unroll
    for (int c = 0; c < CM; c++) a += f[c] * s_wcls[c * NCLASS + o];
    out[(size_t)v * NCLASS + o] = a;
  }
  float t1[CM];
#pragma unroll
  for (int o = 0; o < CM; o++) {
    float a = s_bfe[o];
#pragma unroll
    for (int c = 0; c < CM; c++) a += f[c] * s_wfe[c * CM + o];
    t1[o] = a;
  }
  float* oemb = out + (size_t)n * NCLASS;
#pragma unroll
  for (int o = 0; o < CM; o++) {
    float a = s_ble[o];
#pragma unroll
    for (int c = 0; c < CM; c++) a += t1[c] * s_wle[c * CM + o];
    oemb[(size_t)v * CM + o] = a;
  }
#pragma unroll
  for (int o = 0; o < CM; o++) {
    float a = s_bfr[o];
#pragma unroll
    for (int c = 0; c < CM; c++) a += f[c] * s_wfr[c * CM + o];
    t1[o] = a;
  }
  float s = s_blr;
#pragma unroll
  for (int c = 0; c < CM; c++) s += t1[c] * s_wlr[c];
  out[(size_t)n * (NCLASS + CM) + v] = 1.0f / (1.0f + __expf(-s));
}

extern "C" void kernel_launch(void* const* d_in, const int* in_sizes, int n_in,
                              void* d_out, int out_size, void* d_ws, size_t ws_size,
                              hipStream_t stream) {
  const float* features = (const float*)d_in[0];
  const int*   nbr      = (const int*)d_in[1];
  const float* w0       = (const float*)d_in[2];
  const float* bw       = (const float*)d_in[3];
  const float* bg       = (const float*)d_in[4];
  const float* bb       = (const float*)d_in[5];
  const float* gf       = (const float*)d_in[6];
  const float* bfv      = (const float*)d_in[7];
  const float* wcls = (const float*)d_in[8];  const float* bcls = (const float*)d_in[9];
  const float* wfe  = (const float*)d_in[10]; const float* bfe  = (const float*)d_in[11];
  const float* wle  = (const float*)d_in[12]; const float* ble  = (const float*)d_in[13];
  const float* wfr  = (const float*)d_in[14]; const float* bfr  = (const float*)d_in[15];
  const float* wlr  = (const float*)d_in[16]; const float* blr  = (const float*)d_in[17];
  int n = in_sizes[1] / KK;

  char* wsb = (char*)d_ws;
  size_t cur = 0;
  auto alloc = [&](size_t bytes) -> char* {
    char* p = wsb + cur;
    cur = (cur + bytes + 255) & ~(size_t)255;
    return p;
  };
  float*          featF = (float*)alloc((size_t)n * CM * sizeof(float));
  float*          hF    = (float*)alloc((size_t)n * CM * sizeof(float));
  unsigned short* actB  = (unsigned short*)alloc((size_t)(n + 1) * CM * sizeof(unsigned short));
  int*            nbrT  = (int*)alloc((size_t)n * KK * sizeof(int));
  unsigned short* wpk   = (unsigned short*)alloc((size_t)4 * KK * 2 * 32 * 16 * sizeof(unsigned short));
  float*          part  = (float*)alloc((size_t)NBSTAT * 64 * sizeof(float));
  float*          ss    = (float*)alloc(64 * sizeof(float));
  (void)ws_size; (void)n_in; (void)out_size;

  {
    int tot = 4 * KK * 2 * 32 * 16;
    repack_w_kernel<<<(tot + 255) / 256, 256, 0, stream>>>(bw, wpk);
  }
  nbr_transpose_kernel<<<(n * KK + 255) / 256, 256, 0, stream>>>(nbr, nbrT, n);
  conv0_kernel<<<(n + 127) / 128, 128, 0, stream>>>(features, nbr, w0, featF, n);

  int gridApply = ((n + 1) * (CM / 2) + 255) / 256;
  int gridConv  = (n + 127) / 128;
  size_t convStride = (size_t)KK * 2 * 32 * 16;

  for (int r = 0; r < 2; r++) {
    bn_stats_kernel<<<NBSTAT, 256, 0, stream>>>(featF, part, n);
    bn_finalize_kernel<<<1, 32, 0, stream>>>(part, bg + (r * 2 + 0) * CM, bb + (r * 2 + 0) * CM, ss, n);
    bn_apply_kernel<<<gridApply, 256, 0, stream>>>(featF, ss, (unsigned int*)actB, nullptr, n);
    conv_wmma_kernel<<<gridConv, 256, 0, stream>>>(actB, nbrT, wpk + (size_t)(r * 2 + 0) * convStride,
                                                   nullptr, hF, n);
    bn_stats_kernel<<<NBSTAT, 256, 0, stream>>>(hF, part, n);
    bn_finalize_kernel<<<1, 32, 0, stream>>>(part, bg + (r * 2 + 1) * CM, bb + (r * 2 + 1) * CM, ss, n);
    bn_apply_kernel<<<gridApply, 256, 0, stream>>>(hF, ss, (unsigned int*)actB, nullptr, n);
    conv_wmma_kernel<<<gridConv, 256, 0, stream>>>(actB, nbrT, wpk + (size_t)(r * 2 + 1) * convStride,
                                                   featF, featF, n);
  }
  bn_stats_kernel<<<NBSTAT, 256, 0, stream>>>(featF, part, n);
  bn_finalize_kernel<<<1, 32, 0, stream>>>(part, gf, bfv, ss, n);
  bn_apply_kernel<<<gridApply, 256, 0, stream>>>(featF, ss, (unsigned int*)actB, hF, n);
  heads_kernel<<<(n + 127) / 128, 128, 0, stream>>>(hF, wcls, bcls, wfe, bfe, wle, ble,
                                                    wfr, bfr, wlr, blr, (float*)d_out, n);
}
